// CrossAttention_9423158247687
// MI455X (gfx1250) — matrix operand order repaired and checked by reading, not compiled, not run on hardware
//
#include <hip/hip_runtime.h>
#include <hip/hip_bf16.h>

// ---------------------------------------------------------------------------
// CDNA5 (gfx1250, wave32) cross-attention implementation.
// All matmuls use v_wmma_f32_16x16x32_f16 (f16 operands, f32 accumulate).
// Round 4: attention K/V tiles staged into LDS once per block via the async
// global->LDS path (global_load_async_to_lds_b128 + s_wait_asynccnt), double
// buffered so copies for chunk i+1 overlap compute on chunk i. This removes
// the 8x duplicate K/V loads the 8 waves of a block were issuing.
// ---------------------------------------------------------------------------

typedef _Float16 v8h  __attribute__((ext_vector_type(8)));
typedef _Float16 v16h __attribute__((ext_vector_type(16)));
typedef float    v8f  __attribute__((ext_vector_type(8)));

// ---- WMMA wrapper ----------------------------------------------------------
__device__ __forceinline__ v8f wmma_f16(v16h a, v16h b, v8f c) {
  // (neg_a, A, neg_b, B, c_mod, C, reuse_a, reuse_b)
  return __builtin_amdgcn_wmma_f32_16x16x32_f16(false, a, false, b,
                                                (short)0, c, false, false);
}

// ---- async global -> LDS copy (16B per lane), ASYNCcnt-tracked -------------
__device__ __forceinline__ unsigned lds_addr_of(const void* p) {
  // flat shared pointer = {SHARED_BASE aperture, lds_offset}; HW takes [31:0]
  return (unsigned)(size_t)p;
}
__device__ __forceinline__ void async_b128(unsigned lds_addr, const _Float16* gsrc) {
  asm volatile("global_load_async_to_lds_b128 %0, %1, off"
               :: "v"(lds_addr), "v"(gsrc)
               : "memory");
}
__device__ __forceinline__ void wait_async0() {
#if __has_builtin(__builtin_amdgcn_s_wait_asynccnt)
  __builtin_amdgcn_s_wait_asynccnt(0);
#else
  asm volatile("s_wait_asynccnt 0" ::: "memory");
#endif
}

// ---- Fragment loaders (row-major f16 source, 16B-aligned accesses) ---------
// A operand: 16x32 tile, rows advance by ld.  Per ISA 7.12.2:
//   lanes 0-15  : row M=lane,    K halves {k0..k0+7, k0+16..k0+23}
//   lanes 16-31 : row M=lane-16, K halves {k0+8..k0+15, k0+24..k0+31}
__device__ __forceinline__ v16h load_a_frag(const _Float16* p, int ld, int lane) {
  const _Float16* base = p + (size_t)(lane & 15) * ld + ((lane & 16) ? 8 : 0);
  v8h lo = *(const v8h*)base;
  v8h hi = *(const v8h*)(base + 16);
  v16h r;
#pragma unroll
  for (int i = 0; i < 8; ++i) { r[i] = lo[i]; r[i + 8] = hi[i]; }
  return r;
}

// B operand: 32x16 tile where B[k][n] = src[n][k] (src row-major, stride ld).
//   lanes 0-15  : col n=lane,    K halves {k0..k0+7, k0+16..k0+23}
//   lanes 16-31 : col n=lane-16, K halves {k0+8..k0+15, k0+24..k0+31}
__device__ __forceinline__ v16h load_b_frag(const _Float16* p, int ld, int lane) {
  const _Float16* base = p + (size_t)(lane & 15) * ld + ((lane & 16) ? 8 : 0);
  v8h lo = *(const v8h*)base;
  v8h hi = *(const v8h*)(base + 16);
  v16h r;
#pragma unroll
  for (int i = 0; i < 8; ++i) { r[i] = lo[i]; r[i + 8] = hi[i]; }
  return r;
}

// ---- width-16 cross-lane reductions (wave32; lanes 0-15 / 16-31 groups) ----
__device__ __forceinline__ float rmax16(float v) {
#pragma unroll
  for (int m = 1; m < 16; m <<= 1) v = fmaxf(v, __shfl_xor(v, m, 16));
  return v;
}
__device__ __forceinline__ float rsum16(float v) {
#pragma unroll
  for (int m = 1; m < 16; m <<= 1) v += __shfl_xor(v, m, 16);
  return v;
}

// ---------------------------------------------------------------------------
// LayerNorm (fp32) -> f16 normalized activation, + fused residual copy.
// ---------------------------------------------------------------------------
__global__ __launch_bounds__(256)
void ln_cast_kernel(const float* __restrict__ x, const float* __restrict__ g,
                    const float* __restrict__ beta, _Float16* __restrict__ out,
                    float* __restrict__ resid) {
  __shared__ float s1[256], s2[256];
  const int row = blockIdx.x;
  const float* xr = x + (size_t)row * 1024;
  float v[4], sum = 0.f, sq = 0.f;
#pragma unroll
  for (int i = 0; i < 4; ++i) {
    v[i] = xr[threadIdx.x + 256 * i];
    sum += v[i];
    sq  += v[i] * v[i];
  }
  s1[threadIdx.x] = sum; s2[threadIdx.x] = sq;
  __syncthreads();
  for (int off = 128; off > 0; off >>= 1) {
    if (threadIdx.x < off) {
      s1[threadIdx.x] += s1[threadIdx.x + off];
      s2[threadIdx.x] += s2[threadIdx.x + off];
    }
    __syncthreads();
  }
  const float mu  = s1[0] * (1.f / 1024.f);
  const float var = s2[0] * (1.f / 1024.f) - mu * mu;
  const float inv = rsqrtf(var + 1e-5f);
#pragma unroll
  for (int i = 0; i < 4; ++i) {
    const int c = threadIdx.x + 256 * i;
    const float nv = (v[i] - mu) * inv * g[c] + beta[c];
    out[(size_t)row * 1024 + c]   = (_Float16)nv;
    resid[(size_t)row * 1024 + c] = v[i];
  }
}

// ---------------------------------------------------------------------------
// fp32 -> f16 cast (weights)
// ---------------------------------------------------------------------------
__global__ __launch_bounds__(256)
void cast16_kernel(const float* __restrict__ in, _Float16* __restrict__ out, int n) {
  const int i = blockIdx.x * 256 + threadIdx.x;
  if (i < n) out[i] = (_Float16)in[i];
}

// ---------------------------------------------------------------------------
// Projection GEMM: out[n,o] = act[n,:] . W[o,:] + bias[o]   (both K-contiguous)
// Wave computes 16 rows x 64 cols; K=1024 walked in 2-stage unrolled steps of
// 64 with ping-pong fragment buffers (stage s refills its own buffer for
// stage s+2, so waits stay shallow and loads overlap the matrix pipe).
// ---------------------------------------------------------------------------
__global__ __launch_bounds__(256)
void proj_kernel(const _Float16* __restrict__ act, const _Float16* __restrict__ W,
                 const float* __restrict__ bias, _Float16* __restrict__ out,
                 int nshift, int trans) {
  const int lane = threadIdx.x & 31;
  const int wave = threadIdx.x >> 5;
  const int row0 = blockIdx.x * 128 + wave * 16;
  const int col0 = blockIdx.y * 64;
  const int Nseq = 1 << nshift;

  const _Float16* arow = act + (size_t)row0 * 1024;
  const _Float16* wcol[4];
#pragma unroll
  for (int j = 0; j < 4; ++j) wcol[j] = W + (size_t)(col0 + 16 * j) * 1024;

  // prologue: buffer 0 <- k=0, buffer 1 <- k=32
  v16h a0 = load_a_frag(arow + 0, 1024, lane);
  v16h b0[4];
#pragma unroll
  for (int j = 0; j < 4; ++j) b0[j] = load_b_frag(wcol[j] + 0, 1024, lane);
  v16h a1 = load_a_frag(arow + 32, 1024, lane);
  v16h b1[4];
#pragma unroll
  for (int j = 0; j < 4; ++j) b1[j] = load_b_frag(wcol[j] + 32, 1024, lane);

  v8f acc[4] = {};
  for (int k0 = 0; k0 < 1024; k0 += 64) {
    const int k2 = (k0 + 64) & 1023;   // wraps on final iters (harmless reload)
    const int k3 = (k0 + 96) & 1023;
    const int kp = (k0 + 192) & 1023;
    __builtin_prefetch(arow + (size_t)(lane & 15) * 1024 + kp, 0, 2);

    // ---- stage even: consume buf0 (k0), refill buf0 <- k0+64 ----
#pragma unroll
    for (int j = 0; j < 4; ++j) acc[j] = wmma_f16(a0, b0[j], acc[j]);
    a0 = load_a_frag(arow + k2, 1024, lane);
#pragma unroll
    for (int j = 0; j < 4; ++j) b0[j] = load_b_frag(wcol[j] + k2, 1024, lane);

    // ---- stage odd: consume buf1 (k0+32), refill buf1 <- k0+96 ----
#pragma unroll
    for (int j = 0; j < 4; ++j) acc[j] = wmma_f16(a1, b1[j], acc[j]);
    a1 = load_a_frag(arow + k3, 1024, lane);
#pragma unroll
    for (int j = 0; j < 4; ++j) b1[j] = load_b_frag(wcol[j] + k3, 1024, lane);
  }

  // D layout: lane -> col (lane&15), rows r + 8*(lane>=16)
  const int colL  = lane & 15;
  const int rbase = 8 * ((lane >> 4) & 1);
  float bv[4];
  int   hh[4], dd[4];
#pragma unroll
  for (int j = 0; j < 4; ++j) {
    const int o = col0 + 16 * j + colL;
    bv[j] = bias[o];
    hh[j] = o >> 6;      // head
    dd[j] = o & 63;      // dim within head
  }
#pragma unroll
  for (int r = 0; r < 8; ++r) {
    const int row = row0 + r + rbase;
    const int b   = row >> nshift;
    const int n   = row & (Nseq - 1);
#pragma unroll
    for (int j = 0; j < 4; ++j) {
      const float val = acc[j][r] + bv[j];
      size_t idx;
      if (trans) idx = (((size_t)(b * 16 + hh[j]) * 64 + dd[j]) * Nseq + n);
      else       idx = (((size_t)(b * 16 + hh[j]) * Nseq + n) * 64 + dd[j]);
      out[idx] = (_Float16)val;
    }
  }
}

// ---------------------------------------------------------------------------
// Flash-style cross-attention with async-LDS staged K/V.
//   Q : [b][h][Nq][64] f16        K : [b][h][Nk][64] f16
//   VT: [b][h][64][Nk] f16        Out: [b][Nq][1024] fp32 (heads merged)
// Block = 256 thr (8 waves) all on one (b,h): each 32-key chunk (4KB K tile +
// 4KB VT tile) is copied into LDS ONCE per block via async b128 copies
// (double buffered; copies for chunk i+1 in flight during compute on i).
// Waves read WMMA fragments from LDS (ds_load_b128). Online softmax in fp32.
// ---------------------------------------------------------------------------
__global__ __launch_bounds__(256)
void attn_kernel(const _Float16* __restrict__ Q, const _Float16* __restrict__ K,
                 const _Float16* __restrict__ VT, float* __restrict__ Out,
                 int Nq, int Nk) {
  __shared__ __align__(16) _Float16 Kbuf[2][32 * 64];  // 2 x 4KB, K rows x d
  __shared__ __align__(16) _Float16 Vbuf[2][64 * 32];  // 2 x 4KB, d rows x kv
  __shared__ __align__(16) _Float16 Pbuf[8][16 * 32];  // per-wave P tiles

  const int tid  = threadIdx.x;
  const int lane = tid & 31;
  const int wave = tid >> 5;
  const int bh = blockIdx.y;            // 0..127
  const int b = bh >> 4, h = bh & 15;
  const int q0 = blockIdx.x * 128 + wave * 16;

  const _Float16* qb = Q  + (size_t)bh * Nq * 64;
  const _Float16* kb = K  + (size_t)bh * Nk * 64;
  const _Float16* vb = VT + (size_t)bh * 64 * Nk;

  const v16h qa0 = load_a_frag(qb + (size_t)q0 * 64 + 0,  64, lane);
  const v16h qa1 = load_a_frag(qb + (size_t)q0 * 64 + 32, 64, lane);

  // cooperative async copy of one 32-key chunk: 256 threads x 16B per tile
  const int vd  = tid >> 2;        // VT row (d)       0..63
  const int vsg = tid & 3;         // 16B segment      0..3
  auto copy_chunk = [&](int buf, int kv) {
    // K tile: rows kv..kv+31 are contiguous 4KB in global
    async_b128(lds_addr_of(&Kbuf[buf][tid * 8]),
               kb + (size_t)kv * 64 + tid * 8);
    // VT tile: 64 rows x 64B, global row stride Nk
    async_b128(lds_addr_of(&Vbuf[buf][vd * 32 + vsg * 8]),
               vb + (size_t)vd * Nk + kv + vsg * 8);
  };

  // prologue: stage chunk 0
  copy_chunk(0, 0);
  wait_async0();
  __syncthreads();

  v8f o[4] = {};
  float mrow[8], lrow[8];
#pragma unroll
  for (int r = 0; r < 8; ++r) { mrow[r] = -1e30f; lrow[r] = 0.f; }

  _Float16* myP = &Pbuf[wave][0];
  const int colL  = lane & 15;
  const int rbase = 8 * ((lane >> 4) & 1);
  const float scale = 0.125f;   // 64^-0.5

  for (int kv = 0; kv < Nk; kv += 32) {
    const int cur = (kv >> 5) & 1;
    const int nxt = cur ^ 1;
    const int kn = (kv + 32 < Nk) ? kv + 32 : 0;   // wrap reload is harmless
    const int kp = (kv + 64 < Nk) ? kv + 64 : 0;

    // kick off async staging of the NEXT chunk; prefetch chunk after that
    copy_chunk(nxt, kn);
    __builtin_prefetch(kb + (size_t)(kp + lane) * 64, 0, 2);

    // ---- S = Q . K^T from the LDS K tile ----
    const v16h kf0 = load_b_frag(&Kbuf[cur][0],            64, lane);
    const v16h kf1 = load_b_frag(&Kbuf[cur][32],           64, lane);
    const v16h kf2 = load_b_frag(&Kbuf[cur][16 * 64],      64, lane);
    const v16h kf3 = load_b_frag(&Kbuf[cur][16 * 64 + 32], 64, lane);
    v8f s0 = {}, s1 = {};
    s0 = wmma_f16(qa0, kf0, s0);
    s0 = wmma_f16(qa1, kf1, s0);
    s1 = wmma_f16(qa0, kf2, s1);
    s1 = wmma_f16(qa1, kf3, s1);

    // V fragments from the LDS VT tile (ld = 32)
    v16h vf[4];
#pragma unroll
    for (int j = 0; j < 4; ++j)
      vf[j] = load_b_frag(&Vbuf[cur][(16 * j) * 32], 32, lane);

    // ---- online softmax (fp32 statistics) ----
    float p0[8], p1[8], c[8];
#pragma unroll
    for (int r = 0; r < 8; ++r) {
      const float a  = s0[r] * scale;
      const float bb = s1[r] * scale;
      const float tmax = rmax16(fmaxf(a, bb));
      const float mn = fmaxf(mrow[r], tmax);
      p0[r] = __expf(a - mn);
      p1[r] = __expf(bb - mn);
      const float rs = rsum16(p0[r] + p1[r]);
      c[r] = __expf(mrow[r] - mn);
      lrow[r] = lrow[r] * c[r] + rs;
      mrow[r] = mn;
    }
#pragma unroll
    for (int j = 0; j < 4; ++j)
#pragma unroll
      for (int r = 0; r < 8; ++r) o[j][r] *= c[r];

    // ---- P: D-layout -> LDS (16x32 f16 tile, this wave's region) ----
#pragma unroll
    for (int r = 0; r < 8; ++r) {
      myP[(r + rbase) * 32 + colL]      = (_Float16)p0[r];
      myP[(r + rbase) * 32 + colL + 16] = (_Float16)p1[r];
    }
    __syncthreads();   // P visible; all K/V reads of 'cur' already retired

    // ---- O += P . V ----
    const v16h pa = load_a_frag(myP, 32, lane);
#pragma unroll
    for (int j = 0; j < 4; ++j) o[j] = wmma_f16(pa, vf[j], o[j]);

    // our async copies for 'nxt' must be complete before anyone reads them
    wait_async0();
    __syncthreads();
  }

  // ---- normalize and write merged-head fp32 output ----
#pragma unroll
  for (int j = 0; j < 4; ++j) {
    const int col = h * 64 + 16 * j + colL;
#pragma unroll
    for (int r = 0; r < 8; ++r) {
      const int qrow = q0 + r + rbase;
      Out[((size_t)b * Nq + qrow) * 1024 + col] = o[j][r] / lrow[r];
    }
  }
}

// ---------------------------------------------------------------------------
// Launcher
// ---------------------------------------------------------------------------
extern "C" void kernel_launch(void* const* d_in, const int* in_sizes, int n_in,
                              void* d_out, int out_size, void* d_ws, size_t ws_size,
                              hipStream_t stream) {
  (void)in_sizes; (void)n_in; (void)out_size; (void)ws_size;

  const float* text   = (const float*)d_in[0];   // (8,512,1024)
  const float* vision = (const float*)d_in[1];   // (8,1024,1024)
  const float* n1g = (const float*)d_in[2];
  const float* n1b = (const float*)d_in[3];
  const float* n2g = (const float*)d_in[4];
  const float* n2b = (const float*)d_in[5];
  const float* W[6];
  const float* bias[6];
  for (int i = 0; i < 6; ++i) {            // order: q1,k1,v1,q2,k2,v2
    W[i]    = (const float*)d_in[6 + 2 * i];
    bias[i] = (const float*)d_in[7 + 2 * i];
  }

  float* out = (float*)d_out;
  // output layout (flat, fp32): a1 | a2 | text_residual | vision_residual
  float* a1   = out;                 // 8*512*1024  = 4,194,304
  float* a2   = out + 4194304;       // 8*1024*1024 = 8,388,608
  float* tres = out + 12582912;      // 4,194,304
  float* vres = out + 16777216;      // 8,388,608

  // workspace layout (bytes)
  char* ws = (char*)d_ws;
  _Float16* tnorm = (_Float16*)(ws + 0);          //  8 MB  (4096x1024 f16)
  _Float16* vnorm = (_Float16*)(ws + 8388608);    // 16 MB  (8192x1024 f16)
  _Float16* wf16[6];
  for (int i = 0; i < 6; ++i)
    wf16[i] = (_Float16*)(ws + 25165824 + (size_t)i * 2097152);  // 12 MB total
  _Float16* tq  = (_Float16*)(ws + 37748736);     // [8][16][512][64]
  _Float16* tk  = (_Float16*)(ws + 46137344);
  _Float16* tvT = (_Float16*)(ws + 54525952);     // [8][16][64][512]
  _Float16* vq  = (_Float16*)(ws + 62914560);     // [8][16][1024][64]
  _Float16* vk  = (_Float16*)(ws + 79691776);
  _Float16* vvT = (_Float16*)(ws + 96468992);     // [8][16][64][1024]

  // 1) LayerNorm + f16 cast + fused residual copies
  ln_cast_kernel<<<4096, 256, 0, stream>>>(text,   n1g, n1b, tnorm, tres);
  ln_cast_kernel<<<8192, 256, 0, stream>>>(vision, n2g, n2b, vnorm, vres);

  // 2) weight casts to f16
  for (int i = 0; i < 6; ++i)
    cast16_kernel<<<4096, 256, 0, stream>>>(W[i], wf16[i], 1048576);

  // 3) projections (WMMA GEMMs). text M=4096 (nshift=9), vision M=8192 (10).
  proj_kernel<<<dim3(32, 16), 256, 0, stream>>>(tnorm, wf16[0], bias[0], tq,  9,  0);
  proj_kernel<<<dim3(32, 16), 256, 0, stream>>>(tnorm, wf16[1], bias[1], tk,  9,  0);
  proj_kernel<<<dim3(32, 16), 256, 0, stream>>>(tnorm, wf16[2], bias[2], tvT, 9,  1);
  proj_kernel<<<dim3(64, 16), 256, 0, stream>>>(vnorm, wf16[3], bias[3], vq,  10, 0);
  proj_kernel<<<dim3(64, 16), 256, 0, stream>>>(vnorm, wf16[4], bias[4], vk,  10, 0);
  proj_kernel<<<dim3(64, 16), 256, 0, stream>>>(vnorm, wf16[5], bias[5], vvT, 10, 1);

  // 4) cross attention (flash, WMMA): text Q x vision K/V, vision Q x text K/V
  attn_kernel<<<dim3(4, 128), 256, 0, stream>>>(tq, vk, vvT, a1, 512, 1024);
  attn_kernel<<<dim3(8, 128), 256, 0, stream>>>(vq, tk, tvT, a2, 1024, 512);
}
